// MessagePassingConvolution_49520972923320
// MI455X (gfx1250) — compile-verified, matrix operand order checked by reading, not run on hardware
//
#include <hip/hip_runtime.h>
#include <hip/hip_bf16.h>
#include <stdint.h>

// ---------------------------------------------------------------------------
// Problem constants (from reference)
// ---------------------------------------------------------------------------
#define N_NODES  10000
#define N_EDGES  160000
#define C_CH     32
#define RADIAL   8
#define HID      64
#define OUT_DIM  9           // 1 + 3 + 5
#define NFRAG_G  (64 * 6)    // Wgen fragments: 64 k-slices x 6 N-tiles (N=96)
// workspace layout (bytes)
#define HB_BYTES ((size_t)N_EDGES * HID * 2)              // h[E,64] as bf16
#define WG_OFF   HB_BYTES                                 // Wgen bf16 frags
#define WG_BYTES ((size_t)NFRAG_G * 1024)
#define WL_OFF   (WG_OFF + WG_BYTES)                      // W1/W2/W3 frags (20 frags)

typedef __attribute__((ext_vector_type(16))) __bf16 v16bf;
typedef __attribute__((ext_vector_type(8)))  float  v8f;
typedef __attribute__((ext_vector_type(4)))  unsigned int v4u;
typedef __attribute__((ext_vector_type(8)))  int  v8i;
typedef __attribute__((ext_vector_type(4)))  int  v4i;

union BF16Frag { v16bf v; unsigned int u[8]; uint4 q[2]; };

__device__ __forceinline__ unsigned short f2bf(float f) {
    unsigned int u = __float_as_uint(f);
    u += 0x7FFFu + ((u >> 16) & 1u);   // round-to-nearest-even
    return (unsigned short)(u >> 16);
}
__device__ __forceinline__ unsigned int packbf(float lo, float hi) {
    return (unsigned int)f2bf(lo) | ((unsigned int)f2bf(hi) << 16);
}
__device__ __forceinline__ unsigned int pk_mul_bf16(unsigned int a, unsigned int b) {
    unsigned int d;
    asm volatile("v_pk_mul_bf16 %0, %1, %2" : "=v"(d) : "v"(a), "v"(b));
    return d;
}
// fast silu: one v_exp_f32 + one v_rcp_f32 (bf16-pipeline accuracy is plenty)
__device__ __forceinline__ float silu(float x) {
    return x * __builtin_amdgcn_rcpf(1.0f + __expf(-x));
}
__device__ __forceinline__ v8f zero8() {
    v8f z = {0.f, 0.f, 0.f, 0.f, 0.f, 0.f, 0.f, 0.f};
    return z;
}
// K-index pattern of a 16-bit A/B operand dword slot (d = dword 0..7, h = half,
// q = lane>>4) per CDNA5 ISA 16-bit 16x32 layout.
__device__ __forceinline__ int kpat(int q, int d, int h) {
    return (d < 4) ? (8 * q + 2 * d + h) : (16 + 8 * q + 2 * (d - 4) + h);
}
// low 32 bits of a flat LDS pointer == LDS byte offset (aperture maps addr[31:0])
__device__ __forceinline__ unsigned int lds_off32(const void* p) {
    return (unsigned int)(size_t)p;
}

// ---------------------------------------------------------------------------
// Kernel 0: zero output (harness poisons d_out with 0xAA)
// ---------------------------------------------------------------------------
__global__ void k_zero(float* __restrict__ out) {
    int i = blockIdx.x * 256 + threadIdx.x;
    if (i < N_NODES * C_CH * OUT_DIM) out[i] = 0.0f;
}

// ---------------------------------------------------------------------------
// Kernel 1: pack weights into bf16 WMMA B-operand fragments (scales folded in)
// ---------------------------------------------------------------------------
__global__ void k_pack(const float* __restrict__ Wgen, const float* __restrict__ W1,
                       const float* __restrict__ W2, const float* __restrict__ W3,
                       unsigned short* __restrict__ gfrag, unsigned short* __restrict__ lfrag) {
    int tid = blockIdx.x * 256 + threadIdx.x;
    const int GTOT = NFRAG_G * 512;
    if (tid < GTOT) {
        int f = tid >> 9, wi = tid & 511, lane = wi >> 4, slot = wi & 15;
        int k = f / 6, nt = f % 6, q = lane >> 4, nl = lane & 15, d = slot >> 1, hb = slot & 1;
        int i = kpat(q, d, hb);
        int n = nt * 16 + nl, l = n >> 5, o = n & 31;
        float v = Wgen[((k * 3 + l) * 32 + o) * 32 + i] * 0.022097086912079608f; // 1/sqrt(2048)
        gfrag[tid] = f2bf(v);
    } else if (tid < GTOT + 20 * 512) {
        int t2 = tid - GTOT;
        int f = t2 >> 9, wi = t2 & 511, lane = wi >> 4, slot = wi & 15;
        int q = lane >> 4, nl = lane & 15, d = slot >> 1, hb = slot & 1;
        int kk = kpat(q, d, hb);
        float v;
        if (f < 4) {                       // layer 1, K padded to 32
            int n = f * 16 + nl;
            v = (kk < RADIAL) ? W1[kk * HID + n] * 0.3535533905932738f : 0.0f; // 1/sqrt(8)
        } else if (f < 12) {               // layer 2
            int g = f - 4, kc = g >> 2, nt = g & 3, n = nt * 16 + nl;
            v = W2[(kc * 32 + kk) * HID + n] * 0.125f;                          // 1/sqrt(64)
        } else {                           // layer 3
            int g = f - 12, kc = g >> 2, nt = g & 3, n = nt * 16 + nl;
            v = W3[(kc * 32 + kk) * HID + n] * 0.125f;
        }
        lfrag[t2] = f2bf(v);
    }
}

// ---------------------------------------------------------------------------
// Kernel 2: 3-layer SiLU MLP on radial features via WMMA. One 16-edge tile per
// wave; inter-layer re-layout (C layout -> A layout) through LDS. h out as bf16.
// ---------------------------------------------------------------------------
__global__ __launch_bounds__(256) void k_mlp(const float* __restrict__ efeat,
                                             const unsigned short* __restrict__ lfrag,
                                             unsigned short* __restrict__ hbuf) {
    __shared__ float t_lds[8 * 16 * HID];   // 32 KB: per-wave 16x64 tile
    int tid = threadIdx.x, w = tid >> 5, L = tid & 31, r = L & 15, q = L >> 4;
    float* my = &t_lds[w * 16 * HID];
    int ebase = blockIdx.x * 128 + w * 16;

    // ---- layer 1: A = edge_feats tile, K=8 zero padded to 32 ----
    BF16Frag A;
#pragma unroll
    for (int d = 0; d < 8; ++d) A.u[d] = 0u;
    if (q == 0) {
        const float4* ef = (const float4*)(efeat + (size_t)(ebase + r) * RADIAL);
        float4 f0 = ef[0], f1 = ef[1];
        A.u[0] = packbf(f0.x, f0.y); A.u[1] = packbf(f0.z, f0.w);
        A.u[2] = packbf(f1.x, f1.y); A.u[3] = packbf(f1.z, f1.w);
    }
    v8f acc[4];
#pragma unroll
    for (int nt = 0; nt < 4; ++nt) acc[nt] = zero8();
#pragma unroll
    for (int nt = 0; nt < 4; ++nt) {
        BF16Frag B;
        const uint4* bq = (const uint4*)(lfrag + (size_t)nt * 512 + L * 16);
        B.q[0] = bq[0]; B.q[1] = bq[1];
        acc[nt] = __builtin_amdgcn_wmma_f32_16x16x32_bf16(false, A.v, false, B.v,
                                                          (short)0, acc[nt], false, false);
    }
#pragma unroll
    for (int nt = 0; nt < 4; ++nt)
#pragma unroll
        for (int v = 0; v < 8; ++v)
            my[(v + 8 * q) * HID + nt * 16 + r] = silu(acc[nt][v]);
    __syncthreads();

    // ---- layers 2 and 3 ----
    for (int layer = 0; layer < 2; ++layer) {
        v8f a2[4];
#pragma unroll
        for (int nt = 0; nt < 4; ++nt) a2[nt] = zero8();
#pragma unroll
        for (int kc = 0; kc < 2; ++kc) {
            BF16Frag A2;
#pragma unroll
            for (int d = 0; d < 8; ++d) {
                int kb = kc * 32 + kpat(q, d, 0);
                A2.u[d] = packbf(my[r * HID + kb], my[r * HID + kb + 1]);
            }
#pragma unroll
            for (int nt = 0; nt < 4; ++nt) {
                BF16Frag B;
                const uint4* bq =
                    (const uint4*)(lfrag + (size_t)(4 + layer * 8 + kc * 4 + nt) * 512 + L * 16);
                B.q[0] = bq[0]; B.q[1] = bq[1];
                a2[nt] = __builtin_amdgcn_wmma_f32_16x16x32_bf16(false, A2.v, false, B.v,
                                                                 (short)0, a2[nt], false, false);
            }
        }
        __syncthreads();
        if (layer == 0) {
#pragma unroll
            for (int nt = 0; nt < 4; ++nt)
#pragma unroll
                for (int v = 0; v < 8; ++v)
                    my[(v + 8 * q) * HID + nt * 16 + r] = silu(a2[nt][v]);
            __syncthreads();
        } else {
#pragma unroll
            for (int nt = 0; nt < 4; ++nt)
#pragma unroll
                for (int v = 0; v < 8; ++v)
                    hbuf[(size_t)(ebase + v + 8 * q) * HID + nt * 16 + r] = f2bf(silu(a2[nt][v]));
        }
    }
}

// ---------------------------------------------------------------------------
// Kernel 3: main einsum as WMMA GEMM + message expansion + atomic scatter.
// Block = 256 threads = 8 waves = 256 edges; wave owns 2 M-tiles (32 edges).
// h tile (32 KB) DMA'd in via Tensor Data Mover. B fragments streamed with a
// DOUBLE-BUFFERED GLOBAL_LOAD_ASYNC_TO_LDS pipeline: chunk c+1 is in flight
// while chunk c feeds 24 WMMAs, so the Wgen stream hides behind the matmul.
// ---------------------------------------------------------------------------
__global__ __launch_bounds__(256) void k_main(const float* __restrict__ node_feats,
                                              const float* __restrict__ eattr,
                                              const int* __restrict__ senders,
                                              const int* __restrict__ receivers,
                                              const unsigned short* __restrict__ hbuf,
                                              const unsigned short* __restrict__ gfrag,
                                              float* __restrict__ out) {
    __shared__ unsigned short h_lds[256 * HID];   // 32 KB, bf16 h tile
    __shared__ unsigned int   b_lds[2][3072];     // 2 x 12 KB: 2-k chunks of B frags
    int tid = threadIdx.x, w = tid >> 5, L = tid & 31, r = L & 15, q = L >> 4;
    int e_base = blockIdx.x * 256;

    // ---- TDM: DMA the whole h tile (8192 x 4B) into LDS with one instruction
    if (tid < 32) {
        unsigned long long ga = (unsigned long long)(size_t)(hbuf + (size_t)e_base * HID);
        unsigned int lds0 = lds_off32(&h_lds[0]);
        // D# group0: count=1, lds_addr, global_addr[56:0], type=2
        v4u g0 = { 1u, lds0, (unsigned int)ga,
                   ((unsigned int)((ga >> 32) & 0x1FFFFFFull)) | 0x80000000u };
        // D# group1: data_size=4B (code 2); tensor_dim0 = tile_dim0 = 8192;
        //            tensor_dim0_stride = 8192; everything else 0 (1-D tile)
        v8i g1 = { 0x20000, (int)0x20000000, 0, (int)0x20000000, 0, 8192, 0, 0 };
        v4i g2 = { 0, 0, 0, 0 };
        v4i g3 = { 0, 0, 0, 0 };
        v8i g4 = { 0, 0, 0, 0, 0, 0, 0, 0 };
        __builtin_amdgcn_tensor_load_to_lds(g0, g1, g2, g3, g4, 0);
        __builtin_amdgcn_s_wait_tensorcnt(0);
    }

    // gather sender features, pre-pack A-operand bf16 dwords (unscaled)
    int eT0 = e_base + w * 32 + r, eT1 = eT0 + 16;
    int s0 = senders[eT0], s1 = senders[eT1];
    unsigned int x0[8], x1[8];
    {
        const float4* nf = (const float4*)(node_feats + (size_t)s0 * C_CH);
        float4 a0 = nf[2 * q], a1 = nf[2 * q + 1], b0 = nf[4 + 2 * q], b1 = nf[5 + 2 * q];
        x0[0] = packbf(a0.x, a0.y); x0[1] = packbf(a0.z, a0.w);
        x0[2] = packbf(a1.x, a1.y); x0[3] = packbf(a1.z, a1.w);
        x0[4] = packbf(b0.x, b0.y); x0[5] = packbf(b0.z, b0.w);
        x0[6] = packbf(b1.x, b1.y); x0[7] = packbf(b1.z, b1.w);
    }
    {
        const float4* nf = (const float4*)(node_feats + (size_t)s1 * C_CH);
        float4 a0 = nf[2 * q], a1 = nf[2 * q + 1], b0 = nf[4 + 2 * q], b1 = nf[5 + 2 * q];
        x1[0] = packbf(a0.x, a0.y); x1[1] = packbf(a0.z, a0.w);
        x1[2] = packbf(a1.x, a1.y); x1[3] = packbf(a1.z, a1.w);
        x1[4] = packbf(b0.x, b0.y); x1[5] = packbf(b0.z, b0.w);
        x1[6] = packbf(b1.x, b1.y); x1[7] = packbf(b1.z, b1.w);
    }

    v8f acc[2][6];
#pragma unroll
    for (int t = 0; t < 2; ++t)
#pragma unroll
        for (int nt = 0; nt < 6; ++nt) acc[t][nt] = zero8();

    const unsigned int b_off0 = lds_off32(&b_lds[0][0]) + (unsigned int)tid * 16u;
    const unsigned int b_off1 = lds_off32(&b_lds[1][0]) + (unsigned int)tid * 16u;

    // issue one 2-k chunk (12 KB = 256 threads x 3 x 16B) asynchronously
    auto issue_chunk = [&](int c, unsigned int dst_base) {
        const char* src = (const char*)gfrag + (size_t)c * 12288 + (size_t)tid * 16;
#pragma unroll
        for (int j = 0; j < 3; ++j) {
            asm volatile("global_load_async_to_lds_b128 %0, %1, off"
                         :: "v"(dst_base + (unsigned int)(j * 4096)),
                            "v"(src + (size_t)j * 4096)
                         : "memory");
        }
    };

    issue_chunk(0, b_off0);                  // prologue

    for (int c = 0; c < 32; ++c) {           // 32 chunks of 2 k-values
        const int p = c & 1;
        asm volatile("s_wait_asynccnt 0x0" ::: "memory");  // chunk c resident
        __syncthreads();                     // all waves ready; buffer 1-p free
        if (c + 1 < 32) issue_chunk(c + 1, p ? b_off0 : b_off1);
#pragma unroll
        for (int kk = 0; kk < 2; ++kk) {
            int k = c * 2 + kk;
            unsigned int h0 = h_lds[(w * 32 + r) * HID + k];      h0 |= h0 << 16;
            unsigned int h1 = h_lds[(w * 32 + 16 + r) * HID + k]; h1 |= h1 << 16;
            BF16Frag A0, A1;
#pragma unroll
            for (int d = 0; d < 8; ++d) {
                A0.u[d] = pk_mul_bf16(x0[d], h0);
                A1.u[d] = pk_mul_bf16(x1[d], h1);
            }
#pragma unroll
            for (int nt = 0; nt < 6; ++nt) {
                BF16Frag B;
                const uint4* bq = (const uint4*)&b_lds[p][(kk * 6 + nt) * 256 + L * 8];
                B.q[0] = bq[0]; B.q[1] = bq[1];
                acc[0][nt] = __builtin_amdgcn_wmma_f32_16x16x32_bf16(false, A0.v, false, B.v,
                                                                     (short)0, acc[0][nt], false, false);
                acc[1][nt] = __builtin_amdgcn_wmma_f32_16x16x32_bf16(false, A1.v, false, B.v,
                                                                     (short)0, acc[1][nt], false, false);
            }
        }
    }

    // message expansion + scatter (1/sqrt(16) folded in)
#pragma unroll
    for (int T = 0; T < 2; ++T) {
#pragma unroll
        for (int nt = 0; nt < 6; ++nt) {
            const int l   = nt >> 1;
            const int off = (l == 0) ? 0 : ((l == 1) ? 1 : 4);
            const int dim = (l == 0) ? 1 : ((l == 1) ? 3 : 5);
            const int o   = (nt & 1) * 16 + r;
#pragma unroll
            for (int v = 0; v < 8; ++v) {
                float u = acc[T][nt][v] * 0.25f;
                int e   = e_base + w * 32 + T * 16 + v + 8 * q;
                int rcv = receivers[e];
                float* dst       = out + ((size_t)rcv * C_CH + o) * OUT_DIM + off;
                const float* at  = eattr + (size_t)e * OUT_DIM + off;
#pragma unroll
                for (int j = 0; j < dim; ++j) atomicAdd(dst + j, u * at[j]);
            }
        }
    }
}

// ---------------------------------------------------------------------------
// Host entry
// ---------------------------------------------------------------------------
extern "C" void kernel_launch(void* const* d_in, const int* in_sizes, int n_in,
                              void* d_out, int out_size, void* d_ws, size_t ws_size,
                              hipStream_t stream) {
    const float* node_feats = (const float*)d_in[0];
    const float* edge_attrs = (const float*)d_in[1];
    const float* edge_feats = (const float*)d_in[2];
    const int*   senders    = (const int*)d_in[3];
    const int*   receivers  = (const int*)d_in[4];
    const float* W1         = (const float*)d_in[5];
    const float* W2         = (const float*)d_in[6];
    const float* W3         = (const float*)d_in[7];
    const float* Wgen       = (const float*)d_in[8];
    float* out = (float*)d_out;
    char*  ws  = (char*)d_ws;

    unsigned short* hbuf  = (unsigned short*)ws;
    unsigned short* gfrag = (unsigned short*)(ws + WG_OFF);
    unsigned short* lfrag = (unsigned short*)(ws + WL_OFF);

    k_zero<<<(N_NODES * C_CH * OUT_DIM + 255) / 256, 256, 0, stream>>>(out);
    k_pack<<<(NFRAG_G * 512 + 20 * 512 + 255) / 256, 256, 0, stream>>>(
        Wgen, W1, W2, W3, gfrag, lfrag);
    k_mlp<<<N_EDGES / 128, 256, 0, stream>>>(edge_feats, lfrag, hbuf);
    k_main<<<N_EDGES / 256, 256, 0, stream>>>(node_feats, edge_attrs, senders, receivers,
                                              hbuf, gfrag, out);
}
